// VQTM_60249801228732
// MI455X (gfx1250) — compile-verified, matrix operand order checked by reading, not run on hardware
//
#include <hip/hip_runtime.h>
#include <hip/hip_bf16.h>

typedef __attribute__((ext_vector_type(16))) _Float16 v16h;
typedef __attribute__((ext_vector_type(8)))  _Float16 v8h;
typedef __attribute__((ext_vector_type(8)))  float    v8f;

constexpr int Dd = 512;       // embedding dim
constexpr int Kk = 512;       // num concepts
constexpr int Nn = 131072;    // tokens
constexpr int Vv = 50257;     // vocab
constexpr float COMMIT = 0.25f;
constexpr float PEPS   = 1e-6f;

// ---- d_out layout (flat concat, reference return order) ----
constexpr size_t ENC_OFF  = 0;                               // [N,K]
constexpr size_t QW_OFF   = (size_t)Nn * Kk;                 // [N,D]
constexpr size_t DOCU_OFF = QW_OFF + (size_t)Nn * Dd;        // [1,D]
constexpr size_t OUTS_OFF = DOCU_OFF + Dd;                   // [1,V]
constexpr size_t VQ_OFF   = OUTS_OFF + Vv;                   // scalar
constexpr size_t LTS_OFF  = VQ_OFF + 1;                      // scalar

// ---- workspace layout (bytes) ----
constexpr size_t WS_IDX   = 0;                  // int[N]
constexpr size_t WS_CB16  = 524288;             // half[K*D]
constexpr size_t WS_CNORM = 1048576;            // float[K]
constexpr size_t WS_CSUM  = 1050624;            // float[K]
constexpr size_t WS_HIST  = 1052672;            // float[K]
constexpr size_t WS_BIN   = 1054720;            // float[V]
constexpr size_t WS_LOGIT = 1255936;            // float[V]
constexpr size_t WS_SSEP  = 1457152;            // float[N/8]
constexpr size_t WS_LTSP  = 1522688;            // float[K]
constexpr size_t WS_SCAL  = 1524736;            // float[4]

// ---------- CDNA5 async global->LDS helpers ----------
__device__ __forceinline__ void async_load_b128(const _Float16* gsrc, _Float16* lds_dst) {
  // VDST = 32-bit LDS byte offset, VADDR = 64-bit global address (GV mode).
  const unsigned lds_off = (unsigned)(uintptr_t)lds_dst; // low 32 bits of generic LDS ptr = LDS offset
  asm volatile("global_load_async_to_lds_b128 %0, %1, off"
               :: "v"(lds_off), "v"(gsrc) : "memory");
}

__device__ __forceinline__ void wait_async0() {
#if __has_builtin(__builtin_amdgcn_s_wait_asynccnt)
  __builtin_amdgcn_s_wait_asynccnt(0);
#else
  asm volatile("s_wait_asynccnt 0x0" ::: "memory");
#endif
}

// Zero the accumulator regions we rely on (harness poisons ws with 0xAA once).
__global__ void k_init(float* __restrict__ hist, float* __restrict__ bin) {
  const int i = blockIdx.x * blockDim.x + threadIdx.x;
  if (i < Kk) hist[i] = 0.0f;
  if (i < Vv) bin[i]  = 0.0f;
}

// f16 codebook copy + per-concept norm & sum (warp per concept row)
__global__ void __launch_bounds__(256)
k_prep(const float* __restrict__ cb, _Float16* __restrict__ cb16,
       float* __restrict__ cnorm, float* __restrict__ csum) {
  const int lane = threadIdx.x & 31;
  const int w    = threadIdx.x >> 5;
  const int row  = blockIdx.x * 8 + w;
  const float* cr = cb + (size_t)row * Dd;
  _Float16*    hr = cb16 + (size_t)row * Dd;
  float s = 0.0f, nn = 0.0f;
#pragma unroll
  for (int j = 0; j < 16; ++j) {
    const int d = lane + 32 * j;
    const float v = cr[d];
    s += v; nn += v * v;
    hr[d] = (_Float16)v;
  }
#pragma unroll
  for (int off = 16; off > 0; off >>= 1) {
    s  += __shfl_xor(s,  off, 32);
    nn += __shfl_xor(nn, off, 32);
  }
  if (lane == 0) { cnorm[row] = nn; csum[row] = s; }
}

// ---- Distance GEMM + row argmin -------------------------------------------
// Block = 8 waves x 16 token rows. B tiles (16 concepts x 512) are staged into
// LDS once per block via async global->LDS (double buffered); all 8 waves read
// fragments with ds_load_b128. A fragments live in registers.
// d2' = |c_n|^2 - 2 x_m . c_n  (|x|^2 constant per row -> argmin-invariant).
constexpr int BROW = 520;   // padded row stride (halves): 1040B = 260 dwords = 4 banks mod 64

__global__ void __launch_bounds__(256)
k_gemm_argmin(const int* __restrict__ doc, const float* __restrict__ emb,
              const _Float16* __restrict__ cb16, const float* __restrict__ cnorm,
              int* __restrict__ outIdx) {
  __shared__ __align__(16) _Float16 Bs[2][16][BROW];

  const int tid  = threadIdx.x;
  const int lane = tid & 31;
  const int wave = tid >> 5;
  const int m16  = lane & 15;   // A row / B col / C col within the 16x16 tile
  const int half = lane >> 4;   // selects K sub-range of the fragment
  const int rowBase = blockIdx.x * 128 + wave * 16;

  // ---- build A fragments (16 chunks of K=32) in registers, fp32 -> f16 ----
  // ISA 16-bit A layout: lanes 0-15 hold K = k0+half*8 .. +7 and k0+16+half*8 .. +7
  v16h afrag[16];
  {
    const float* xr = emb + (size_t)doc[rowBase + m16] * Dd;
#pragma unroll
    for (int kc = 0; kc < 16; ++kc) {
      const int k0 = kc * 32;
      const float* p0 = xr + k0 + half * 8;
      const float* p1 = xr + k0 + 16 + half * 8;
      const float4 a0 = *(const float4*)(p0);
      const float4 a1 = *(const float4*)(p0 + 4);
      const float4 a2 = *(const float4*)(p1);
      const float4 a3 = *(const float4*)(p1 + 4);
      v16h a;
      a[0]=(_Float16)a0.x;  a[1]=(_Float16)a0.y;  a[2]=(_Float16)a0.z;  a[3]=(_Float16)a0.w;
      a[4]=(_Float16)a1.x;  a[5]=(_Float16)a1.y;  a[6]=(_Float16)a1.z;  a[7]=(_Float16)a1.w;
      a[8]=(_Float16)a2.x;  a[9]=(_Float16)a2.y;  a[10]=(_Float16)a2.z; a[11]=(_Float16)a2.w;
      a[12]=(_Float16)a3.x; a[13]=(_Float16)a3.y; a[14]=(_Float16)a3.z; a[15]=(_Float16)a3.w;
      afrag[kc] = a;
    }
  }

  // ---- stage tile 0 (16KB = 1024 x 16B segments; 4 per thread) ----
  // segment s: row = s>>6, 16B-chunk c16 = s&63
#pragma unroll
  for (int j = 0; j < 4; ++j) {
    const int s   = tid + j * 256;
    const int row = s >> 6;
    const int c16 = s & 63;
    async_load_b128(cb16 + (size_t)row * Dd + c16 * 8, &Bs[0][row][c16 * 8]);
  }

  float bestVal[8];
  int   bestIdx[8];
#pragma unroll
  for (int r = 0; r < 8; ++r) { bestVal[r] = 3.0e38f; bestIdx[r] = 0; }

  for (int t = 0; t < Kk / 16; ++t) {
    wait_async0();        // this wave's stage of tile t has landed in LDS
    __syncthreads();      // all waves' stages visible; safe to overwrite other buffer

    if (t + 1 < Kk / 16) {
      const int n1 = (t + 1) * 16;
      _Float16* buf = &Bs[(t + 1) & 1][0][0];
#pragma unroll
      for (int j = 0; j < 4; ++j) {
        const int s   = tid + j * 256;
        const int row = s >> 6;
        const int c16 = s & 63;
        async_load_b128(cb16 + (size_t)(n1 + row) * Dd + c16 * 8,
                        buf + row * BROW + c16 * 8);
      }
    }

    const int n0 = t * 16;
    // B = C^T : fragment for column n = n0+m16 is two contiguous 16B LDS reads
    const _Float16* brow = &Bs[t & 1][m16][0];
    v8f acc = {};
#pragma unroll
    for (int kc = 0; kc < 16; ++kc) {
      const int k0 = kc * 32;
      const v8h b0 = *(const v8h*)(brow + k0 + half * 8);
      const v8h b1 = *(const v8h*)(brow + k0 + 16 + half * 8);
      v16h b;
#pragma unroll
      for (int i = 0; i < 8; ++i) { b[i] = b0[i]; b[i + 8] = b1[i]; }
      acc = __builtin_amdgcn_wmma_f32_16x16x32_f16(
          false, afrag[kc], false, b, (short)0, acc, false, false);
    }
    // C/D layout: VGPR r, lane -> row M = r + 8*half, col N = n0 + m16
    const float cn = cnorm[n0 + m16];
#pragma unroll
    for (int r = 0; r < 8; ++r) {
      const float d2 = cn - 2.0f * acc[r];
      if (d2 < bestVal[r]) { bestVal[r] = d2; bestIdx[r] = n0 + m16; }
    }
  }
  // min-reduce across the 16 lanes holding one row (xor stays within half)
#pragma unroll
  for (int off = 1; off < 16; off <<= 1) {
#pragma unroll
    for (int r = 0; r < 8; ++r) {
      const float ov = __shfl_xor(bestVal[r], off, 32);
      const int   oi = __shfl_xor(bestIdx[r], off, 32);
      if (ov < bestVal[r] || (ov == bestVal[r] && oi < bestIdx[r])) {
        bestVal[r] = ov; bestIdx[r] = oi;
      }
    }
  }
  if (m16 == 0) {
#pragma unroll
    for (int r = 0; r < 8; ++r)
      outIdx[rowBase + r + 8 * half] = bestIdx[r];
  }
}

// One-hot encodings + quantized rows + SSE partials + code histogram (warp/token)
__global__ void __launch_bounds__(256)
k_assign(const int* __restrict__ doc, const float* __restrict__ emb,
         const float* __restrict__ cb, const int* __restrict__ idx,
         float* __restrict__ enc, float* __restrict__ qw,
         float* __restrict__ hist, float* __restrict__ ssePart) {
  __shared__ float ps[8];
  const int lane = threadIdx.x & 31;
  const int w    = threadIdx.x >> 5;
  const int n    = blockIdx.x * 8 + w;
  const int k    = idx[n];
  const float* crow = cb  + (size_t)k * Dd;
  const float* xrow = emb + (size_t)doc[n] * Dd;
  float* qrow = qw  + (size_t)n * Dd;
  float* erow = enc + (size_t)n * Kk;
  float sse = 0.0f;
#pragma unroll
  for (int j = 0; j < 4; ++j) {
    const int d = lane * 4 + j * 128;
    const float4 q = *(const float4*)(crow + d);
    const float4 x = *(const float4*)(xrow + d);
    *(float4*)(qrow + d) = q;
    const float dx = q.x - x.x, dy = q.y - x.y, dz = q.z - x.z, dw2 = q.w - x.w;
    sse += dx * dx + dy * dy + dz * dz + dw2 * dw2;
    float4 e;
    e.x = (d + 0 == k) ? 1.0f : 0.0f;
    e.y = (d + 1 == k) ? 1.0f : 0.0f;
    e.z = (d + 2 == k) ? 1.0f : 0.0f;
    e.w = (d + 3 == k) ? 1.0f : 0.0f;
    *(float4*)(erow + d) = e;
  }
#pragma unroll
  for (int off = 16; off > 0; off >>= 1) sse += __shfl_xor(sse, off, 32);
  if (lane == 0) { ps[w] = sse; atomicAdd(&hist[k], 1.0f); }
  __syncthreads();
  if (threadIdx.x == 0) {
    float t = 0.0f;
#pragma unroll
    for (int i = 0; i < 8; ++i) t += ps[i];
    ssePart[blockIdx.x] = t;
  }
}

__global__ void k_bincount(const int* __restrict__ doc, float* __restrict__ bin) {
  const int i = blockIdx.x * blockDim.x + threadIdx.x;
  if (i < Nn) atomicAdd(&bin[doc[i]], 1.0f);
}

// quantized_docu[d] = sum_k hist[k]*cb[k][d] / N
__global__ void __launch_bounds__(512)
k_docu(const float* __restrict__ cb, const float* __restrict__ hist,
       float* __restrict__ docu) {
  const int d = threadIdx.x;
  float acc = 0.0f;
  for (int k = 0; k < Kk; ++k) acc += hist[k] * cb[(size_t)k * Dd + d];
  docu[d] = acc * (1.0f / (float)Nn);
}

// Hinge loss over KxK pairwise distances (block per row i)
__global__ void __launch_bounds__(256)
k_lts(const float* __restrict__ cb, const float* __restrict__ cnorm,
      const float* __restrict__ csum, float* __restrict__ ltsPart) {
  __shared__ float ci[Dd];
  __shared__ float red[256];
  const int i = blockIdx.x;
  for (int d = threadIdx.x; d < Dd; d += 256) ci[d] = cb[(size_t)i * Dd + d];
  __syncthreads();
  const float ni = cnorm[i], si = csum[i];
  float acc = 0.0f;
  for (int j = threadIdx.x; j < Kk; j += 256) {
    const float* cj = cb + (size_t)j * Dd;
    float dot = 0.0f;
    for (int d = 0; d < Dd; ++d) dot += ci[d] * cj[d];
    const float dd = ni + cnorm[j] - 2.0f * dot
                   + 2.0f * PEPS * (si - csum[j]) + (float)Dd * PEPS * PEPS;
    const float dist = sqrtf(fmaxf(dd, 0.0f));
    acc += (j == i) ? dist : fmaxf(0.0f, 1.0f - dist);
  }
  red[threadIdx.x] = acc;
  __syncthreads();
  for (int off = 128; off > 0; off >>= 1) {
    if (threadIdx.x < off) red[threadIdx.x] += red[threadIdx.x + off];
    __syncthreads();
  }
  if (threadIdx.x == 0) ltsPart[i] = red[0];
}

// logits[v] = docu . q2v_weight[v] + bias[v]  (warp per vocab row)
__global__ void __launch_bounds__(256)
k_logits(const float* __restrict__ docu, const float* __restrict__ w2,
         const float* __restrict__ bias, float* __restrict__ logits) {
  const int lane = threadIdx.x & 31;
  const int w    = threadIdx.x >> 5;
  const int v    = blockIdx.x * 8 + w;
  if (v >= Vv) return;
  const float* wr = w2 + (size_t)v * Dd;
  float acc = 0.0f;
#pragma unroll
  for (int j = 0; j < 16; ++j) { const int d = lane + 32 * j; acc += docu[d] * wr[d]; }
#pragma unroll
  for (int off = 16; off > 0; off >>= 1) acc += __shfl_xor(acc, off, 32);
  if (lane == 0) logits[v] = acc + bias[v];
}

// single-block max + sum(exp) over V (deterministic fixed-tree)
__global__ void __launch_bounds__(256)
k_smreduce(const float* __restrict__ logits, float* __restrict__ scal) {
  __shared__ float red[256];
  float m = -3.0e38f;
  for (int v = threadIdx.x; v < Vv; v += 256) m = fmaxf(m, logits[v]);
  red[threadIdx.x] = m; __syncthreads();
  for (int off = 128; off > 0; off >>= 1) {
    if (threadIdx.x < off) red[threadIdx.x] = fmaxf(red[threadIdx.x], red[threadIdx.x + off]);
    __syncthreads();
  }
  const float mx = red[0];
  __syncthreads();
  float s = 0.0f;
  for (int v = threadIdx.x; v < Vv; v += 256) s += expf(logits[v] - mx);
  red[threadIdx.x] = s; __syncthreads();
  for (int off = 128; off > 0; off >>= 1) {
    if (threadIdx.x < off) red[threadIdx.x] += red[threadIdx.x + off];
    __syncthreads();
  }
  if (threadIdx.x == 0) { scal[0] = mx; scal[1] = red[0]; }
}

__global__ void k_outputs(const float* __restrict__ logits, const float* __restrict__ bin,
                          const float* __restrict__ scal, float* __restrict__ out) {
  const int v = blockIdx.x * blockDim.x + threadIdx.x;
  if (v >= Vv) return;
  const float p = expf(logits[v] - scal[0]) / scal[1];
  out[v] = logf(p + 1e-6f) * bin[v];
}

__global__ void __launch_bounds__(256)
k_final(const float* __restrict__ ssePart, const float* __restrict__ ltsPart,
        float* __restrict__ outVq, float* __restrict__ outLts) {
  __shared__ float red[256];
  float s = 0.0f;
  for (int i = threadIdx.x; i < Nn / 8; i += 256) s += ssePart[i];
  red[threadIdx.x] = s; __syncthreads();
  for (int off = 128; off > 0; off >>= 1) {
    if (threadIdx.x < off) red[threadIdx.x] += red[threadIdx.x + off];
    __syncthreads();
  }
  if (threadIdx.x == 0)
    *outVq = (1.0f + COMMIT) * red[0] / ((float)Nn * (float)Dd);
  __syncthreads();
  float l = 0.0f;
  for (int i = threadIdx.x; i < Kk; i += 256) l += ltsPart[i];
  red[threadIdx.x] = l; __syncthreads();
  for (int off = 128; off > 0; off >>= 1) {
    if (threadIdx.x < off) red[threadIdx.x] += red[threadIdx.x + off];
    __syncthreads();
  }
  if (threadIdx.x == 0)
    *outLts = red[0] / ((float)Kk * (float)Kk);
}

extern "C" void kernel_launch(void* const* d_in, const int* in_sizes, int n_in,
                              void* d_out, int out_size, void* d_ws, size_t ws_size,
                              hipStream_t stream) {
  (void)in_sizes; (void)n_in; (void)out_size; (void)ws_size;
  const int*   doc  = (const int*)d_in[0];
  const float* emb  = (const float*)d_in[1];
  const float* cb   = (const float*)d_in[2];
  const float* w2   = (const float*)d_in[3];
  const float* bias = (const float*)d_in[4];
  float* out = (float*)d_out;

  char* ws = (char*)d_ws;
  int*      idx    = (int*)(ws + WS_IDX);
  _Float16* cb16   = (_Float16*)(ws + WS_CB16);
  float*    cnorm  = (float*)(ws + WS_CNORM);
  float*    csum   = (float*)(ws + WS_CSUM);
  float*    hist   = (float*)(ws + WS_HIST);
  float*    bin    = (float*)(ws + WS_BIN);
  float*    logits = (float*)(ws + WS_LOGIT);
  float*    ssePart= (float*)(ws + WS_SSEP);
  float*    ltsPart= (float*)(ws + WS_LTSP);
  float*    scal   = (float*)(ws + WS_SCAL);

  k_init<<<(Vv + 255) / 256, 256, 0, stream>>>(hist, bin);
  k_prep<<<Kk / 8, 256, 0, stream>>>(cb, cb16, cnorm, csum);
  k_gemm_argmin<<<Nn / 128, 256, 0, stream>>>(doc, emb, cb16, cnorm, idx);
  k_assign<<<Nn / 8, 256, 0, stream>>>(doc, emb, cb, idx,
                                       out + ENC_OFF, out + QW_OFF, hist, ssePart);
  k_bincount<<<(Nn + 255) / 256, 256, 0, stream>>>(doc, bin);
  k_docu<<<1, 512, 0, stream>>>(cb, hist, out + DOCU_OFF);
  k_lts<<<Kk, 256, 0, stream>>>(cb, cnorm, csum, ltsPart);
  k_logits<<<(Vv + 7) / 8, 256, 0, stream>>>(out + DOCU_OFF, w2, bias, logits);
  k_smreduce<<<1, 256, 0, stream>>>(logits, scal);
  k_outputs<<<(Vv + 255) / 256, 256, 0, stream>>>(logits, bin, scal, out + OUTS_OFF);
  k_final<<<1, 256, 0, stream>>>(ssePart, ltsPart, out + VQ_OFF, out + LTS_OFF);
}
